// GAT_34557306863750
// MI455X (gfx1250) — matrix operand order repaired and checked by reading, not compiled, not run on hardware
//
#include <hip/hip_runtime.h>
#include <hip/hip_bf16.h>
#include <stdint.h>

// ---------------------------------------------------------------------------
// Sizes from the reference
// ---------------------------------------------------------------------------
#define N_NODES 20000
#define E_PER_REL 100000
#define E_TOT (E_PER_REL + N_NODES)   // edges + self loops
#define IN_DIM 512
#define NHEADS 2
#define C1 512
#define HC1 (NHEADS * C1)             // 1024
#define C2 128
#define HC2 (NHEADS * C2)             // 256
#define NREL 6
#define NEG_SLOPE 0.2f

typedef __attribute__((ext_vector_type(16))) __bf16 v16bf;
typedef __attribute__((ext_vector_type(8)))  __bf16 v8bf;
typedef __attribute__((ext_vector_type(8)))  float  v8f;

struct Rels { const int* e[NREL]; };

// ---------------------------------------------------------------------------
// CDNA5 async global->LDS copy (ASYNCcnt-tracked DMA, no VGPR staging).
// VDST operand is the wave-relative LDS byte offset; the LDS aperture's flat
// base has zero low-32 bits, so truncating a generic LDS pointer yields it.
// GVS addressing: 64-bit SGPR base + 32-bit per-lane unsigned byte offset.
// ---------------------------------------------------------------------------
__device__ __forceinline__ void async_copy_b128(const void* sbase, unsigned gofs, void* lds_ptr) {
    unsigned l = (unsigned)(uintptr_t)lds_ptr;
    asm volatile("global_load_async_to_lds_b128 %0, %1, %2"
                 :: "v"(l), "v"(gofs), "s"((unsigned long long)(uintptr_t)sbase)
                 : "memory");
}
#define WAIT_ASYNCCNT(n) asm volatile("s_wait_asynccnt %0" :: "n"(n) : "memory")

// ---------------------------------------------------------------------------
// Elementwise helpers
// ---------------------------------------------------------------------------
__global__ void cvt_f32_bf16(const float* __restrict__ in, __bf16* __restrict__ out, int n) {
    int i = blockIdx.x * blockDim.x + threadIdx.x;
    if (i < n) out[i] = (__bf16)in[i];
}

__global__ void fill_f32(float* __restrict__ p, float v, int n) {
    int i = blockIdx.x * blockDim.x + threadIdx.x;
    if (i < n) p[i] = v;
}

// h = relu(acc + 6*bias) -> bf16 (6 relations each add bias in the reference)
__global__ void bias_relu_bf16(const float* __restrict__ acc, const float* __restrict__ bias,
                               __bf16* __restrict__ out, int hc, int n) {
    int i = blockIdx.x * blockDim.x + threadIdx.x;
    if (i < n) {
        float v = acc[i] + 6.0f * bias[i % hc];
        out[i] = (__bf16)(v > 0.0f ? v : 0.0f);
    }
}

__global__ void add_bias6(float* __restrict__ out, const float* __restrict__ bias, int hc, int n) {
    int i = blockIdx.x * blockDim.x + threadIdx.x;
    if (i < n) out[i] += 6.0f * bias[i % hc];
}

// ---------------------------------------------------------------------------
// bf16 WMMA GEMM:  Y[M,N] = A[M,K] * W[N,K]^T   (all row-major)
// Block = 256 threads (8 waves, wave32). Block tile 128x128, wave tile 32x64
// (8x v_wmma_f32_16x16x32_bf16 per K-step per wave), K step 32.
// Double-buffered LDS filled by global_load_async_to_lds_b128.
// NOTE: no row guard on A loads: out-of-range rows only feed accumulator rows
// that are never stored, and reads stay inside the workspace carve.
// ---------------------------------------------------------------------------
#define BM 128
#define BN 128
#define BK 32

__global__ __launch_bounds__(256)
void gemm_bf16_wmma(const __bf16* __restrict__ A, const __bf16* __restrict__ W,
                    float* __restrict__ Y, int M, int N, int K) {
    __shared__ __bf16 As[2][BM][BK];   // 16 KB
    __shared__ __bf16 Ws[2][BN][BK];   // 16 KB

    const int tid  = threadIdx.x;
    const int wave = tid >> 5;
    const int lane = tid & 31;
    const int wr   = wave >> 1;     // wave row 0..3  (32 rows each)
    const int wc   = wave & 1;      // wave col 0..1  (64 cols each)
    const int m0   = blockIdx.y * BM;
    const int n0   = blockIdx.x * BN;
    const int hl   = lane >> 4;     // lane half
    const int l16  = lane & 15;

    v8f acc[2][4] = {};

    // Issue one K-slice of async DMA into LDS buffer b (4 ops per thread/wave).
    auto issue = [&](int b, int k0) {
        #pragma unroll
        for (int q = tid; q < (BM * BK) / 8; q += 256) {
            int row  = q >> 2;
            int col8 = (q & 3) * 8;
            async_copy_b128(A, (unsigned)(((size_t)(m0 + row) * K + k0 + col8) * 2),
                            &As[b][row][col8]);
            async_copy_b128(W, (unsigned)(((size_t)(n0 + row) * K + k0 + col8) * 2),
                            &Ws[b][row][col8]);
        }
    };

    const int nsteps = K / BK;
    issue(0, 0);

    for (int s = 0; s < nsteps; ++s) {
        const int b = s & 1;
        if (s + 1 < nsteps) {            // prefetch next slice into other buffer
            issue(b ^ 1, (s + 1) * BK);
            WAIT_ASYNCCNT(4);            // retire this buffer's 4 older DMAs
        } else {
            WAIT_ASYNCCNT(0);
        }
        __syncthreads();

        // Fragment loads per CDNA5 16-bit WMMA layouts.
        v16bf afrag[2], bfrag[4];
        #pragma unroll
        for (int t = 0; t < 2; ++t) {
            // A 16x32: lanes 0-15 row M=l16 hold K {0..7,16..23}; lanes 16-31 hold {8..15,24..31}
            int am = wr * 32 + t * 16 + l16;
            const __bf16* ap = &As[b][am][hl * 8];
            v8bf alo = *(const v8bf*)ap;          // K = hl*8 .. hl*8+7
            v8bf ahi = *(const v8bf*)(ap + 16);   // K = 16+hl*8 ..
            v16bf a;
            #pragma unroll
            for (int i = 0; i < 8; ++i) { a[i] = alo[i]; a[i + 8] = ahi[i]; }
            afrag[t] = a;
        }
        #pragma unroll
        for (int t = 0; t < 4; ++t) {
            // B 32x16: lanes 0-15 col N=l16 hold K {0..7,16..23}; lanes 16-31 hold {8..15,24..31}
            int bn = wc * 64 + t * 16 + l16;
            const __bf16* bp = &Ws[b][bn][hl * 8];
            v8bf blo = *(const v8bf*)bp;
            v8bf bhi = *(const v8bf*)(bp + 16);
            v16bf bb;
            #pragma unroll
            for (int i = 0; i < 8; ++i) { bb[i] = blo[i]; bb[i + 8] = bhi[i]; }
            bfrag[t] = bb;
        }

        #pragma unroll
        for (int tr = 0; tr < 2; ++tr)
            #pragma unroll
            for (int tc = 0; tc < 4; ++tc)
                acc[tr][tc] = __builtin_amdgcn_wmma_f32_16x16x32_bf16(
                    false, afrag[tr], false, bfrag[tc],
                    (short)0, acc[tr][tc], false, false);
        __syncthreads();   // protect buffer b before next iteration refills it
    }

    // Store: C/D 16x16 f32 layout: lane half hl, VGPR j -> row M = hl*8 + j, col N = l16
    #pragma unroll
    for (int tr = 0; tr < 2; ++tr) {
        #pragma unroll
        for (int tc = 0; tc < 4; ++tc) {
            int cn = n0 + wc * 64 + tc * 16 + l16;
            #pragma unroll
            for (int j = 0; j < 8; ++j) {
                int cm = m0 + wr * 32 + tr * 16 + hl * 8 + j;
                if (cm < M) Y[(size_t)cm * N + cn] = acc[tr][tc][j];
            }
        }
    }
}

// ---------------------------------------------------------------------------
// al[n,h] = sum_c X[n, h*C + c] * a[h, c]     (one block per node)
// ---------------------------------------------------------------------------
__global__ void head_logits(const float* __restrict__ X, const float* __restrict__ a,
                            float* __restrict__ al, int C, int HC) {
    int n = blockIdx.x;
    __shared__ float sums[NHEADS];
    if (threadIdx.x < NHEADS) sums[threadIdx.x] = 0.0f;
    __syncthreads();
    float p0 = 0.0f, p1 = 0.0f;
    const float* xr = X + (size_t)n * HC;
    for (int c = threadIdx.x; c < HC; c += blockDim.x) {
        float v = xr[c] * a[c];
        if (c < C) p0 += v; else p1 += v;
    }
    atomicAdd(&sums[0], p0);
    atomicAdd(&sums[1], p1);
    __syncthreads();
    if (threadIdx.x < NHEADS) al[n * NHEADS + threadIdx.x] = sums[threadIdx.x];
}

// ---------------------------------------------------------------------------
// Edge passes, all 6 relations at once (blockIdx.y = relation).
// Edge id t in [0, E_TOT): t < E -> real edge, else self-loop node t-E.
// e is [2][E] (row 0 src, row 1 dst).
// ---------------------------------------------------------------------------
__device__ __forceinline__ void atomicMaxF32(float* addr, float val) {
    if (!(val < 0.0f)) atomicMax((int*)addr, __float_as_int(val));
    else               atomicMin((unsigned int*)addr, __float_as_uint(val));
}

__global__ void edge_max_all(Rels rels,
                             const float* __restrict__ als, const float* __restrict__ ald,
                             float* __restrict__ ebuf, float* __restrict__ mbuf) {
    int rel = blockIdx.y;
    const int* e = rels.e[rel];
    float* eb = ebuf + (size_t)rel * E_TOT * NHEADS;
    float* mb = mbuf + (size_t)rel * N_NODES * NHEADS;
    int i = blockIdx.x * blockDim.x + threadIdx.x;
    if (i >= E_TOT * NHEADS) return;
    int t = i >> 1, h = i & 1;
    int src, dst;
    if (t < E_PER_REL) { src = e[t]; dst = e[E_PER_REL + t]; }
    else               { src = dst = t - E_PER_REL; }
    float v = als[src * NHEADS + h] + ald[dst * NHEADS + h];
    v = v > 0.0f ? v : NEG_SLOPE * v;          // leaky_relu
    eb[i] = v;
    atomicMaxF32(&mb[dst * NHEADS + h], v);
}

__global__ void edge_expsum_all(Rels rels,
                                float* __restrict__ ebuf,  // in: logits, out: p
                                const float* __restrict__ mbuf, float* __restrict__ sbuf) {
    int rel = blockIdx.y;
    const int* e = rels.e[rel];
    float*       eb = ebuf + (size_t)rel * E_TOT * NHEADS;
    const float* mb = mbuf + (size_t)rel * N_NODES * NHEADS;
    float*       sb = sbuf + (size_t)rel * N_NODES * NHEADS;
    int i = blockIdx.x * blockDim.x + threadIdx.x;
    if (i >= E_TOT * NHEADS) return;
    int t = i >> 1, h = i & 1;
    int dst = (t < E_PER_REL) ? e[E_PER_REL + t] : (t - E_PER_REL);
    float p = __expf(eb[i] - mb[dst * NHEADS + h]);
    eb[i] = p;
    atomicAdd(&sb[dst * NHEADS + h], p);
}

__global__ void edge_scatter_all(Rels rels,
                                 const float* __restrict__ ebuf, const float* __restrict__ sbuf,
                                 const float* __restrict__ xs, float* __restrict__ out,
                                 int HC, int C) {
    int rel = blockIdx.y;
    const int* e = rels.e[rel];
    const float* pb = ebuf + (size_t)rel * E_TOT * NHEADS;
    const float* sb = sbuf + (size_t)rel * N_NODES * NHEADS;
    int t = blockIdx.x;                   // one block per edge
    int src, dst;
    if (t < E_PER_REL) { src = e[t]; dst = e[E_PER_REL + t]; }
    else               { src = dst = t - E_PER_REL; }
    __shared__ float alpha[NHEADS];
    if (threadIdx.x < NHEADS) {
        float pp = pb[(size_t)t * NHEADS + threadIdx.x];
        float ss = sb[(size_t)dst * NHEADS + threadIdx.x];
        alpha[threadIdx.x] = pp / (ss + 1e-16f);
    }
    __syncthreads();
    const float* xr = xs + (size_t)src * HC;
    float* orow = out + (size_t)dst * HC;
    for (int c = threadIdx.x; c < HC; c += blockDim.x) {
        float a = alpha[c >= C];          // H == 2
        atomicAdd(&orow[c], xr[c] * a);
    }
}

// ---------------------------------------------------------------------------
// Launch
// ---------------------------------------------------------------------------
static inline int ceil_div(int a, int b) { return (a + b - 1) / b; }

extern "C" void kernel_launch(void* const* d_in, const int* in_sizes, int n_in,
                              void* d_out, int out_size, void* d_ws, size_t ws_size,
                              hipStream_t stream) {
    const float* x      = (const float*)d_in[0];
    Rels rels;
    for (int r = 0; r < NREL; ++r) rels.e[r] = (const int*)d_in[1 + r];
    const float* W1_src = (const float*)d_in[7];
    const float* W1_dst = (const float*)d_in[8];
    const float* a1_src = (const float*)d_in[9];
    const float* a1_dst = (const float*)d_in[10];
    const float* b1     = (const float*)d_in[11];
    const float* W2_src = (const float*)d_in[12];
    const float* W2_dst = (const float*)d_in[13];
    const float* a2_src = (const float*)d_in[14];
    const float* a2_dst = (const float*)d_in[15];
    const float* b2     = (const float*)d_in[16];
    float* out = (float*)d_out;

    // ---- workspace carving (all sizes 256B-aligned) -----------------------
    char* ws = (char*)d_ws;
    size_t off = 0;
    auto carve = [&](size_t bytes) { char* p = ws + off; off += (bytes + 255) & ~(size_t)255; return p; };

    __bf16* x_bf   = (__bf16*)carve((size_t)N_NODES * IN_DIM * 2);   // 20.5 MB
    __bf16* w1s_bf = (__bf16*)carve((size_t)HC1 * IN_DIM * 2);       //  1.0 MB
    __bf16* w1d_bf = (__bf16*)carve((size_t)HC1 * IN_DIM * 2);
    __bf16* w2s_bf = (__bf16*)carve((size_t)HC2 * HC1 * 2);          //  0.5 MB
    __bf16* w2d_bf = (__bf16*)carve((size_t)HC2 * HC1 * 2);
    float*  bufA   = (float*)carve((size_t)N_NODES * HC1 * 4);       // 82 MB: xs1, then h_bf16
    float*  bufB   = (float*)carve((size_t)N_NODES * HC1 * 4);       // 82 MB: xd1, then acc1, then xs2+xd2
    float*  als    = (float*)carve((size_t)N_NODES * NHEADS * 4);
    float*  ald    = (float*)carve((size_t)N_NODES * NHEADS * 4);
    float*  mbuf   = (float*)carve((size_t)NREL * N_NODES * NHEADS * 4);  // 1 MB
    float*  sbuf   = (float*)carve((size_t)NREL * N_NODES * NHEADS * 4);  // 1 MB
    float*  ebuf   = (float*)carve((size_t)NREL * E_TOT * NHEADS * 4);    // 5.8 MB
    (void)ws_size; (void)n_in; (void)in_sizes; (void)out_size;

    float*  xs1   = bufA;
    float*  xd1   = bufB;
    float*  acc1  = bufB;                               // reuse (xd1 dead after logits)
    __bf16* h_bf  = (__bf16*)bufA;                      // reuse (xs1 dead after layer-1 scatter)
    float*  xs2   = bufB;                               // reuse (acc1 dead after bias_relu)
    float*  xd2   = bufB + (size_t)N_NODES * HC2;

    const int TB = 256;
    const int nEdgeTh = E_TOT * NHEADS;
    const int nMS = NREL * N_NODES * NHEADS;

    // ---- convert inputs to bf16 ------------------------------------------
    cvt_f32_bf16<<<ceil_div(N_NODES * IN_DIM, TB), TB, 0, stream>>>(x, x_bf, N_NODES * IN_DIM);
    cvt_f32_bf16<<<ceil_div(HC1 * IN_DIM, TB), TB, 0, stream>>>(W1_src, w1s_bf, HC1 * IN_DIM);
    cvt_f32_bf16<<<ceil_div(HC1 * IN_DIM, TB), TB, 0, stream>>>(W1_dst, w1d_bf, HC1 * IN_DIM);
    cvt_f32_bf16<<<ceil_div(HC2 * HC1, TB), TB, 0, stream>>>(W2_src, w2s_bf, HC2 * HC1);
    cvt_f32_bf16<<<ceil_div(HC2 * HC1, TB), TB, 0, stream>>>(W2_dst, w2d_bf, HC2 * HC1);

    // ---- layer 1 ----------------------------------------------------------
    {
        dim3 g(HC1 / BN, ceil_div(N_NODES, BM));
        gemm_bf16_wmma<<<g, 256, 0, stream>>>(x_bf, w1s_bf, xs1, N_NODES, HC1, IN_DIM);
        gemm_bf16_wmma<<<g, 256, 0, stream>>>(x_bf, w1d_bf, xd1, N_NODES, HC1, IN_DIM);
    }
    head_logits<<<N_NODES, TB, 0, stream>>>(xs1, a1_src, als, C1, HC1);
    head_logits<<<N_NODES, TB, 0, stream>>>(xd1, a1_dst, ald, C1, HC1);
    fill_f32<<<ceil_div(N_NODES * HC1, TB), TB, 0, stream>>>(acc1, 0.0f, N_NODES * HC1);
    fill_f32<<<ceil_div(nMS, TB), TB, 0, stream>>>(mbuf, -__builtin_huge_valf(), nMS);
    fill_f32<<<ceil_div(nMS, TB), TB, 0, stream>>>(sbuf, 0.0f, nMS);
    {
        dim3 ge(ceil_div(nEdgeTh, TB), NREL);
        edge_max_all<<<ge, TB, 0, stream>>>(rels, als, ald, ebuf, mbuf);
        edge_expsum_all<<<ge, TB, 0, stream>>>(rels, ebuf, mbuf, sbuf);
        dim3 gs(E_TOT, NREL);
        edge_scatter_all<<<gs, TB, 0, stream>>>(rels, ebuf, sbuf, xs1, acc1, HC1, C1);
    }
    bias_relu_bf16<<<ceil_div(N_NODES * HC1, TB), TB, 0, stream>>>(acc1, b1, h_bf, HC1, N_NODES * HC1);

    // ---- layer 2 ----------------------------------------------------------
    {
        dim3 g(HC2 / BN, ceil_div(N_NODES, BM));
        gemm_bf16_wmma<<<g, 256, 0, stream>>>(h_bf, w2s_bf, xs2, N_NODES, HC2, HC1);
        gemm_bf16_wmma<<<g, 256, 0, stream>>>(h_bf, w2d_bf, xd2, N_NODES, HC2, HC1);
    }
    head_logits<<<N_NODES, TB, 0, stream>>>(xs2, a2_src, als, C2, HC2);
    head_logits<<<N_NODES, TB, 0, stream>>>(xd2, a2_dst, ald, C2, HC2);
    fill_f32<<<ceil_div(N_NODES * HC2, TB), TB, 0, stream>>>(out, 0.0f, N_NODES * HC2);
    fill_f32<<<ceil_div(nMS, TB), TB, 0, stream>>>(mbuf, -__builtin_huge_valf(), nMS);
    fill_f32<<<ceil_div(nMS, TB), TB, 0, stream>>>(sbuf, 0.0f, nMS);
    {
        dim3 ge(ceil_div(nEdgeTh, TB), NREL);
        edge_max_all<<<ge, TB, 0, stream>>>(rels, als, ald, ebuf, mbuf);
        edge_expsum_all<<<ge, TB, 0, stream>>>(rels, ebuf, mbuf, sbuf);
        dim3 gs(E_TOT, NREL);
        edge_scatter_all<<<gs, TB, 0, stream>>>(rels, ebuf, sbuf, xs2, out, HC2, C2);
    }
    add_bias6<<<ceil_div(N_NODES * HC2, TB), TB, 0, stream>>>(out, b2, HC2, N_NODES * HC2);
}
